// InvariantPointAttention_15564961481080
// MI455X (gfx1250) — compile-verified, matrix operand order checked by reading, not run on hardware
//
#include <hip/hip_runtime.h>
#include <hip/hip_bf16.h>
#include <math.h>

typedef __attribute__((ext_vector_type(16))) _Float16 v16h;
typedef __attribute__((ext_vector_type(8)))  _Float16 v8h;
typedef __attribute__((ext_vector_type(8)))  float    v8f;
typedef __attribute__((ext_vector_type(4)))  unsigned int u32x4;
typedef __attribute__((ext_vector_type(4)))  int      i32x4;
typedef __attribute__((ext_vector_type(8)))  int      i32x8;

#define L_SEQ   768
#define DN_DIM  256
#define DP_DIM  128
#define H_HEADS 8
#define DH_DIM  128
#define NQP     4
#define NVP     8
#define PROJ_N  3456     // 1024(Q) + 1024(K) + 1024(V) + 96(Qp) + 96(Kp) + 192(Vp)
#define QCAT_D  160      // 128 + 12 + 2 bias feats, padded to multiple of 32
#define OUT_CAT 2304     // H*(Dh + Nvp*3 + Nvp + DP)

static constexpr float SCALEF = 0.13867504905630728f;   // (16 + 36)^-0.5
static constexpr float WCF    = 0.23570226039551587f;   // sqrt(2/36)

#define HAVE_TDM __has_builtin(__builtin_amdgcn_tensor_load_to_lds)

// ---------------------------------------------------------------------------
// WMMA fragment load: 16-bit A (16x32 MxK) / B (32x16 KxN stored as N x K rows).
// lane 0..15  -> row (lane),    K = k0+0..7  (elems 0..7), K = k0+16..23 (8..15)
// lane 16..31 -> row (lane-16), K = k0+8..15 (elems 0..7), K = k0+24..31 (8..15)
// ---------------------------------------------------------------------------
__device__ __forceinline__ v16h load_frag16(const _Float16* __restrict__ base,
                                            int ld, int r0, int k0, int lane) {
  int r  = r0 + (lane & 15);
  int kb = k0 + ((lane & 16) ? 8 : 0);
  const _Float16* p = base + (size_t)r * (size_t)ld + kb;
  v8h lo = *(const v8h*)(p);
  v8h hi = *(const v8h*)(p + 16);
  v16h f;
#pragma unroll
  for (int i = 0; i < 8; ++i) { f[i] = lo[i]; f[i + 8] = hi[i]; }
  return f;
}

// Generic f16 WMMA GEMM: C[M x N] (fp32) = A[M x K] * Bt[N x K]^T
// Each wave computes one 16x16 tile; blockIdx.y indexes a batch (head) with
// per-batch element strides sA/sB/sC. mode==1: C = acc + addin[m*ldc+n] + bias[n].
__global__ void gemm_f16_wmma(const _Float16* __restrict__ A,
                              const _Float16* __restrict__ Bt,
                              float* __restrict__ C,
                              int M, int N, int K,
                              int lda, int ldb, int ldc,
                              long long sA, long long sB, long long sC,
                              int mode,
                              const float* __restrict__ addin,
                              const float* __restrict__ bias) {
  int wave = threadIdx.x >> 5;
  int lane = threadIdx.x & 31;
  int tiles_n = N >> 4;
  int tiles_m = M >> 4;
  int tile = blockIdx.x * (blockDim.x >> 5) + wave;
  if (tile >= tiles_m * tiles_n) return;
  int tm = tile / tiles_n;
  int tn = tile - tm * tiles_n;
  const _Float16* Ab = A + (long long)blockIdx.y * sA;
  const _Float16* Bb = Bt + (long long)blockIdx.y * sB;
  float* Cb = C + (long long)blockIdx.y * sC;
  int m0 = tm << 4, n0 = tn << 4;
  v8f acc = {0.f, 0.f, 0.f, 0.f, 0.f, 0.f, 0.f, 0.f};
  for (int k = 0; k < K; k += 32) {
    v16h a = load_frag16(Ab, lda, m0, k, lane);
    v16h b = load_frag16(Bb, ldb, n0, k, lane);
    acc = __builtin_amdgcn_wmma_f32_16x16x32_f16(false, a, false, b,
                                                 (short)0, acc, false, false);
  }
  int n = n0 + (lane & 15);
  int mbase = m0 + ((lane & 16) ? 8 : 0);
#pragma unroll
  for (int e = 0; e < 8; ++e) {
    int m = mbase + e;
    float v = acc[e];
    if (mode == 1) v += addin[(size_t)m * ldc + n] + bias[n];
    Cb[(size_t)m * ldc + n] = v;
  }
}

// ---------------------------------------------------------------------------
__global__ void ln_kernel(const float* __restrict__ x, const float* __restrict__ g,
                          const float* __restrict__ b, _Float16* __restrict__ xo) {
  __shared__ float red[DN_DIM];
  int l = blockIdx.x, t = threadIdx.x;
  float v = x[(size_t)l * DN_DIM + t];
  red[t] = v; __syncthreads();
  for (int s = DN_DIM / 2; s > 0; s >>= 1) {
    if (t < s) red[t] += red[t + s];
    __syncthreads();
  }
  float mu = red[0] * (1.0f / DN_DIM); __syncthreads();
  float d = v - mu;
  red[t] = d * d; __syncthreads();
  for (int s = DN_DIM / 2; s > 0; s >>= 1) {
    if (t < s) red[t] += red[t + s];
    __syncthreads();
  }
  float var = red[0] * (1.0f / DN_DIM);
  xo[(size_t)l * DN_DIM + t] = (_Float16)(d * rsqrtf(var + 1e-5f) * g[t] + b[t]);
}

// W is K x ncols (row-major); Wt is ncols x K (row-major), fp32 -> f16.
__global__ void convert_wt(const float* __restrict__ W, _Float16* __restrict__ Wt,
                           int ncols, int K) {
  int idx = blockIdx.x * blockDim.x + threadIdx.x;
  if (idx >= ncols * K) return;
  int n = idx / K, k = idx - n * K;
  Wt[(size_t)n * K + k] = (_Float16)W[(size_t)k * ncols + n];
}

__global__ void convert_f16(const float* __restrict__ src, _Float16* __restrict__ dst, int n) {
  int i = blockIdx.x * blockDim.x + threadIdx.x;
  if (i < n) dst[i] = (_Float16)src[i];
}

// ---------------------------------------------------------------------------
// Pack: rotations + augmented Q/K features + transposed V / Vp_g (f16).
__global__ void pack_kernel(const float* __restrict__ proj, const float* __restrict__ T,
                            _Float16* __restrict__ Qcat, _Float16* __restrict__ Kcat,
                            _Float16* __restrict__ Vt, _Float16* __restrict__ Vpgt) {
  int id = blockIdx.x * blockDim.x + threadIdx.x;
  if (id >= L_SEQ * H_HEADS) return;
  int l = id >> 3, h = id & 7;
  const float* row = proj + (size_t)l * PROJ_N;
  float R[3][3];
#pragma unroll
  for (int i = 0; i < 3; ++i)
#pragma unroll
    for (int j = 0; j < 3; ++j) R[i][j] = T[l * 16 + i * 4 + j];

  _Float16* q  = Qcat + ((size_t)h * L_SEQ + l) * QCAT_D;
  _Float16* kc = Kcat + ((size_t)h * L_SEQ + l) * QCAT_D;
  for (int d = 0; d < DH_DIM; ++d) {
    q[d]  = (_Float16)(row[h * DH_DIM + d] * SCALEF);
    kc[d] = (_Float16)(row[1024 + h * DH_DIM + d]);
  }
  float qsq = 0.f, ksq = 0.f;
#pragma unroll
  for (int p = 0; p < NQP; ++p) {
    float a0 = row[3072 + h * 12 + p * 3 + 0];
    float a1 = row[3072 + h * 12 + p * 3 + 1];
    float a2 = row[3072 + h * 12 + p * 3 + 2];
    float g0 = R[0][0] * a0 + R[0][1] * a1 + R[0][2] * a2;
    float g1 = R[1][0] * a0 + R[1][1] * a1 + R[1][2] * a2;
    float g2 = R[2][0] * a0 + R[2][1] * a1 + R[2][2] * a2;
    qsq += g0 * g0 + g1 * g1 + g2 * g2;
    q[128 + p * 3 + 0] = (_Float16)(g0 * WCF);
    q[128 + p * 3 + 1] = (_Float16)(g1 * WCF);
    q[128 + p * 3 + 2] = (_Float16)(g2 * WCF);
    float b0 = row[3168 + h * 12 + p * 3 + 0];
    float b1 = row[3168 + h * 12 + p * 3 + 1];
    float b2 = row[3168 + h * 12 + p * 3 + 2];
    float k0 = R[0][0] * b0 + R[0][1] * b1 + R[0][2] * b2;
    float k1 = R[1][0] * b0 + R[1][1] * b1 + R[1][2] * b2;
    float k2 = R[2][0] * b0 + R[2][1] * b1 + R[2][2] * b2;
    ksq += k0 * k0 + k1 * k1 + k2 * k2;
    kc[128 + p * 3 + 0] = (_Float16)k0;
    kc[128 + p * 3 + 1] = (_Float16)k1;
    kc[128 + p * 3 + 2] = (_Float16)k2;
  }
  q[140]  = (_Float16)(-0.5f * WCF * qsq);
  q[141]  = (_Float16)1.0f;
  kc[140] = (_Float16)1.0f;
  kc[141] = (_Float16)(-0.5f * WCF * ksq);
#pragma unroll
  for (int i = 142; i < QCAT_D; ++i) { q[i] = (_Float16)0.f; kc[i] = (_Float16)0.f; }

  for (int d = 0; d < DH_DIM; ++d)
    Vt[((size_t)h * DH_DIM + d) * L_SEQ + l] = (_Float16)row[2048 + h * DH_DIM + d];

#pragma unroll
  for (int v = 0; v < NVP; ++v) {
    float a0 = row[3264 + h * 24 + v * 3 + 0];
    float a1 = row[3264 + h * 24 + v * 3 + 1];
    float a2 = row[3264 + h * 24 + v * 3 + 2];
    float g0 = R[0][0] * a0 + R[0][1] * a1 + R[0][2] * a2;
    float g1 = R[1][0] * a0 + R[1][1] * a1 + R[1][2] * a2;
    float g2 = R[2][0] * a0 + R[2][1] * a1 + R[2][2] * a2;
    Vpgt[((size_t)h * 32 + v * 3 + 0) * L_SEQ + l] = (_Float16)g0;
    Vpgt[((size_t)h * 32 + v * 3 + 1) * L_SEQ + l] = (_Float16)g1;
    Vpgt[((size_t)h * 32 + v * 3 + 2) * L_SEQ + l] = (_Float16)g2;
  }
#pragma unroll
  for (int i = 24; i < 32; ++i)
    Vpgt[((size_t)h * 32 + i) * L_SEQ + l] = (_Float16)0.f;
}

// logits[h][l][m] = softplus(head_w[h]) * logits[h][l][m] + pair[l][m][:] . Wpb[:,h]
__global__ void pb_kernel(const float* __restrict__ pair, const float* __restrict__ Wpb,
                          const float* __restrict__ head_w, float* __restrict__ logits) {
  __shared__ float sw[DP_DIM * H_HEADS];
  __shared__ float shw[H_HEADS];
  int t = threadIdx.x;
  for (int i = t; i < DP_DIM * H_HEADS; i += blockDim.x) sw[i] = Wpb[i];
  if (t < H_HEADS) {
    float x = head_w[t];
    shw[t] = fmaxf(x, 0.f) + log1pf(expf(-fabsf(x)));   // softplus
  }
  __syncthreads();
  int idx = blockIdx.x * blockDim.x + t;
  if (idx >= L_SEQ * L_SEQ) return;
  int l = idx / L_SEQ, m = idx - l * L_SEQ;
  const float* p = pair + ((size_t)l * L_SEQ + m) * DP_DIM;
  float acc[H_HEADS];
#pragma unroll
  for (int h = 0; h < H_HEADS; ++h) acc[h] = 0.f;
  for (int d = 0; d < DP_DIM; ++d) {
    float pv = p[d];
#pragma unroll
    for (int h = 0; h < H_HEADS; ++h) acc[h] += pv * sw[d * H_HEADS + h];
  }
#pragma unroll
  for (int h = 0; h < H_HEADS; ++h) {
    size_t o = ((size_t)h * L_SEQ + l) * L_SEQ + m;
    logits[o] = shw[h] * logits[o] + acc[h];
  }
}

__global__ void softmax_kernel(const float* __restrict__ logits, _Float16* __restrict__ attn) {
  __shared__ float red[256];
  int row = blockIdx.x, t = threadIdx.x;
  const float* src = logits + (size_t)row * L_SEQ;
  float x0 = src[t], x1 = src[t + 256], x2 = src[t + 512];
  float mx = fmaxf(x0, fmaxf(x1, x2));
  red[t] = mx; __syncthreads();
  for (int s = 128; s > 0; s >>= 1) {
    if (t < s) red[t] = fmaxf(red[t], red[t + s]);
    __syncthreads();
  }
  mx = red[0]; __syncthreads();
  float e0 = expf(x0 - mx), e1 = expf(x1 - mx), e2 = expf(x2 - mx);
  red[t] = e0 + e1 + e2; __syncthreads();
  for (int s = 128; s > 0; s >>= 1) {
    if (t < s) red[t] += red[t + s];
    __syncthreads();
  }
  float inv = 1.0f / red[0];
  _Float16* dst = attn + (size_t)row * L_SEQ;
  dst[t]       = (_Float16)(e0 * inv);
  dst[t + 256] = (_Float16)(e1 * inv);
  dst[t + 512] = (_Float16)(e2 * inv);
}

// ---------------------------------------------------------------------------
// out_pair[l][h][d] = sum_m attn[h][l][m] * pair[l][m][d]
// The 8x768 f16 attn slice (row stride 1.18MB) is staged into LDS by the
// Tensor Data Mover: one TENSOR_LOAD_TO_LDS with a 2D D# (tile 384x8 dwords,
// dim0_stride = 294912 dwords), synchronized via s_wait_tensorcnt + barrier.
// ---------------------------------------------------------------------------
__global__ void outpair_kernel(const _Float16* __restrict__ attn, const float* __restrict__ pair,
                               float* __restrict__ outcat) {
  __shared__ _Float16 saH[H_HEADS * L_SEQ];   // 12 KB (TDM destination)
  __shared__ float    saF[H_HEADS][L_SEQ];    // 24 KB (compute copy)
  int l = blockIdx.x, t = threadIdx.x;        // blockDim = 128, t = d

#if HAVE_TDM
  if (t < 32) {  // one wave issues the TDM descriptor (TDM ignores EXEC)
    unsigned long long ga =
        (unsigned long long)(uintptr_t)(attn + (size_t)l * L_SEQ);
    unsigned lds_off = (unsigned)(uintptr_t)(&saH[0]);  // low 32 bits = LDS byte offset
    const unsigned rowU32    = (L_SEQ * 2) / 4;                          // 384
    const unsigned strideU32 = (unsigned)((size_t)L_SEQ * L_SEQ * 2 / 4); // 294912

    u32x4 g0;
    g0[0] = 1u;                                        // count=1, user D#
    g0[1] = lds_off;                                   // lds_addr
    g0[2] = (unsigned)(ga & 0xffffffffu);              // global_addr[31:0]
    g0[3] = (unsigned)((ga >> 32) & 0x01ffffffu) | (2u << 30); // addr[56:32] | type=2

    i32x8 g1;
    g1[0] = (int)(2u << 16);                           // wg_mask=0, data_size=4B
    g1[1] = (int)((rowU32 & 0xffffu) << 16);           // tensor_dim0[15:0] @bit48
    g1[2] = (int)(((rowU32 >> 16) & 0xffffu)           // tensor_dim0[31:16]
          |       ((unsigned)H_HEADS << 16));          // tensor_dim1[15:0] @bit80
    g1[3] = (int)((rowU32 & 0xffffu) << 16);           // tensor_dim1[31:16]=0, tile_dim0
    g1[4] = (int)(H_HEADS & 0xffffu);                  // tile_dim1=8, tile_dim2=0
    g1[5] = (int)strideU32;                            // tensor_dim0_stride[31:0]
    g1[6] = 0;                                         // stride0[47:32]=0, stride1 lo=0
    g1[7] = 0;                                         // tensor_dim1_stride hi
    i32x4 gz = {0, 0, 0, 0};
#if __clang_major__ >= 23
    i32x8 gz8 = {0, 0, 0, 0, 0, 0, 0, 0};
    __builtin_amdgcn_tensor_load_to_lds(g0, g1, gz, gz, gz8, 0);
#else
    __builtin_amdgcn_tensor_load_to_lds(g0, g1, gz, gz, 0);
#endif
#if __has_builtin(__builtin_amdgcn_s_wait_tensorcnt)
    __builtin_amdgcn_s_wait_tensorcnt(0);
#endif
  }
  __syncthreads();
  for (int idx = t; idx < H_HEADS * L_SEQ; idx += blockDim.x)
    saF[idx / L_SEQ][idx - (idx / L_SEQ) * L_SEQ] = (float)saH[idx];
  __syncthreads();
#else
  for (int idx = t; idx < H_HEADS * L_SEQ; idx += blockDim.x) {
    int h = idx / L_SEQ, m = idx - h * L_SEQ;
    saF[h][m] = (float)attn[((size_t)h * L_SEQ + l) * L_SEQ + m];
  }
  __syncthreads();
#endif

  float acc[H_HEADS];
#pragma unroll
  for (int h = 0; h < H_HEADS; ++h) acc[h] = 0.f;
  const float* pl = pair + (size_t)l * L_SEQ * DP_DIM + t;
  for (int m = 0; m < L_SEQ; ++m) {
    if ((m & 31) == 0)
      __builtin_prefetch(pl + (size_t)(m + 32) * DP_DIM, 0, 1);
    float pv = pl[(size_t)m * DP_DIM];
#pragma unroll
    for (int h = 0; h < H_HEADS; ++h) acc[h] += saF[h][m] * pv;
  }
  float* dst = outcat + (size_t)l * OUT_CAT + 1280 + t;
#pragma unroll
  for (int h = 0; h < H_HEADS; ++h) dst[h * DH_DIM] = acc[h];
}

// out_p_local = R^T * out_p; norms; write into concat buffer.
__global__ void finalize_kernel(const float* __restrict__ outp_tmp, const float* __restrict__ T,
                                float* __restrict__ outcat) {
  int id = blockIdx.x * blockDim.x + threadIdx.x;
  if (id >= L_SEQ * H_HEADS) return;
  int l = id >> 3, h = id & 7;
  float R[3][3];
#pragma unroll
  for (int i = 0; i < 3; ++i)
#pragma unroll
    for (int j = 0; j < 3; ++j) R[i][j] = T[l * 16 + i * 4 + j];
  const float* src = outp_tmp + (size_t)l * 256 + h * 32;
  float* dl = outcat + (size_t)l * OUT_CAT + 1024 + h * 24;
  float* dn = outcat + (size_t)l * OUT_CAT + 1216 + h * 8;
#pragma unroll
  for (int v = 0; v < NVP; ++v) {
    float o0 = src[v * 3 + 0], o1 = src[v * 3 + 1], o2 = src[v * 3 + 2];
    float l0 = R[0][0] * o0 + R[1][0] * o1 + R[2][0] * o2;
    float l1 = R[0][1] * o0 + R[1][1] * o1 + R[2][1] * o2;
    float l2 = R[0][2] * o0 + R[1][2] * o1 + R[2][2] * o2;
    dl[v * 3 + 0] = l0; dl[v * 3 + 1] = l1; dl[v * 3 + 2] = l2;
    dn[v] = sqrtf(l0 * l0 + l1 * l1 + l2 * l2);
  }
}

// ---------------------------------------------------------------------------
extern "C" void kernel_launch(void* const* d_in, const int* in_sizes, int n_in,
                              void* d_out, int out_size, void* d_ws, size_t ws_size,
                              hipStream_t stream) {
  (void)in_sizes; (void)n_in; (void)out_size; (void)ws_size;
  const float* single = (const float*)d_in[0];
  const float* pair   = (const float*)d_in[1];
  const float* T      = (const float*)d_in[2];
  const float* ln_g   = (const float*)d_in[3];
  const float* ln_b   = (const float*)d_in[4];
  const float* Wq     = (const float*)d_in[5];
  const float* Wk     = (const float*)d_in[6];
  const float* Wv     = (const float*)d_in[7];
  const float* Wqp    = (const float*)d_in[8];
  const float* Wkp    = (const float*)d_in[9];
  const float* Wvp    = (const float*)d_in[10];
  const float* Wpb    = (const float*)d_in[11];
  const float* head_w = (const float*)d_in[12];
  const float* Wout   = (const float*)d_in[13];
  const float* bout   = (const float*)d_in[14];
  float* out = (float*)d_out;

  char* ws = (char*)d_ws;
  size_t off = 0;
  auto alloc = [&](size_t bytes) -> void* {
    void* p = ws + off;
    off = (off + bytes + 255) & ~(size_t)255;
    return p;
  };
  _Float16* xf     = (_Float16*)alloc((size_t)L_SEQ * DN_DIM * 2);
  _Float16* WcatT  = (_Float16*)alloc((size_t)PROJ_N * DN_DIM * 2);
  _Float16* WoutT  = (_Float16*)alloc((size_t)DN_DIM * OUT_CAT * 2);
  float*    proj   = (float*)   alloc((size_t)L_SEQ * PROJ_N * 4);
  _Float16* Qcat   = (_Float16*)alloc((size_t)H_HEADS * L_SEQ * QCAT_D * 2);
  _Float16* Kcat   = (_Float16*)alloc((size_t)H_HEADS * L_SEQ * QCAT_D * 2);
  _Float16* Vt     = (_Float16*)alloc((size_t)H_HEADS * DH_DIM * L_SEQ * 2);
  _Float16* Vpgt   = (_Float16*)alloc((size_t)H_HEADS * 32 * L_SEQ * 2);
  float*    logits = (float*)   alloc((size_t)H_HEADS * L_SEQ * L_SEQ * 4);
  _Float16* attn   = (_Float16*)alloc((size_t)H_HEADS * L_SEQ * L_SEQ * 2);
  float*    outp   = (float*)   alloc((size_t)L_SEQ * 256 * 4);
  float*    ocat   = (float*)   alloc((size_t)L_SEQ * OUT_CAT * 4);
  _Float16* ocatH  = (_Float16*)alloc((size_t)L_SEQ * OUT_CAT * 2);

  // 1) layernorm -> f16
  ln_kernel<<<L_SEQ, DN_DIM, 0, stream>>>(single, ln_g, ln_b, xf);

  // 2) weights -> f16, transposed (N x K)
  auto cvt = [&](const float* W, _Float16* dst, int ncols) {
    int n = ncols * DN_DIM;
    convert_wt<<<(n + 255) / 256, 256, 0, stream>>>(W, dst, ncols, DN_DIM);
  };
  cvt(Wq,  WcatT + (size_t)0    * DN_DIM, 1024);
  cvt(Wk,  WcatT + (size_t)1024 * DN_DIM, 1024);
  cvt(Wv,  WcatT + (size_t)2048 * DN_DIM, 1024);
  cvt(Wqp, WcatT + (size_t)3072 * DN_DIM, 96);
  cvt(Wkp, WcatT + (size_t)3168 * DN_DIM, 96);
  cvt(Wvp, WcatT + (size_t)3264 * DN_DIM, 192);
  {
    int n = DN_DIM * OUT_CAT;   // Wout is 2304 x 256 -> WoutT 256 x 2304
    convert_wt<<<(n + 255) / 256, 256, 0, stream>>>(Wout, WoutT, DN_DIM, OUT_CAT);
  }

  // 3) all projections in one WMMA GEMM: proj(768x3456) = xf @ Wcat
  {
    int tiles = (L_SEQ / 16) * (PROJ_N / 16);
    gemm_f16_wmma<<<dim3((tiles + 3) / 4, 1), 128, 0, stream>>>(
        xf, WcatT, proj, L_SEQ, PROJ_N, DN_DIM,
        DN_DIM, DN_DIM, PROJ_N, 0, 0, 0, 0, nullptr, nullptr);
  }

  // 4) rotations + augmented features + transposed V/Vp_g
  pack_kernel<<<(L_SEQ * H_HEADS + 255) / 256, 256, 0, stream>>>(proj, T, Qcat, Kcat, Vt, Vpgt);

  // 5a) logits = Qcat @ Kcat^T per head (= attn_s + attn_p via augmented feats)
  {
    int tiles = (L_SEQ / 16) * (L_SEQ / 16);
    gemm_f16_wmma<<<dim3((tiles + 3) / 4, H_HEADS), 128, 0, stream>>>(
        Qcat, Kcat, logits, L_SEQ, L_SEQ, QCAT_D,
        QCAT_D, QCAT_D, L_SEQ,
        (long long)L_SEQ * QCAT_D, (long long)L_SEQ * QCAT_D,
        (long long)L_SEQ * L_SEQ, 0, nullptr, nullptr);
  }
  // 5b) logits = softplus(head_w)*logits + pair@Wpb   (streams pair, pass 1)
  pb_kernel<<<(L_SEQ * L_SEQ) / 256, 256, 0, stream>>>(pair, Wpb, head_w, logits);

  // 6) softmax -> f16 attn
  softmax_kernel<<<H_HEADS * L_SEQ, 256, 0, stream>>>(logits, attn);

  // 7a) out_s = attn @ V  (WMMA, per head) -> ocat[:, h*128 + d]
  {
    int tiles = (L_SEQ / 16) * (DH_DIM / 16);
    gemm_f16_wmma<<<dim3((tiles + 3) / 4, H_HEADS), 128, 0, stream>>>(
        attn, Vt, ocat, L_SEQ, DH_DIM, L_SEQ,
        L_SEQ, L_SEQ, OUT_CAT,
        (long long)L_SEQ * L_SEQ, (long long)DH_DIM * L_SEQ, (long long)DH_DIM,
        0, nullptr, nullptr);
  }
  // 7b) out_p = attn @ Vp_g (N padded to 32)
  {
    int tiles = (L_SEQ / 16) * (32 / 16);
    gemm_f16_wmma<<<dim3((tiles + 3) / 4, H_HEADS), 128, 0, stream>>>(
        attn, Vpgt, outp, L_SEQ, 32, L_SEQ,
        L_SEQ, L_SEQ, 256,
        (long long)L_SEQ * L_SEQ, (long long)32 * L_SEQ, (long long)32,
        0, nullptr, nullptr);
  }
  // 7c) out_pair = attn @ pair   (streams pair, pass 2; TDM-staged attn)
  outpair_kernel<<<L_SEQ, 128, 0, stream>>>(attn, pair, ocat);

  // 8) inverse rotation + norms
  finalize_kernel<<<(L_SEQ * H_HEADS + 255) / 256, 256, 0, stream>>>(outp, T, ocat);

  // 9) concat -> f16
  {
    int n = L_SEQ * OUT_CAT;
    convert_f16<<<(n + 255) / 256, 256, 0, stream>>>(ocat, ocatH, n);
  }
  // 10) out = single + ocat @ Wout + bout  (WMMA + fused epilogue)
  {
    int tiles = (L_SEQ / 16) * (DN_DIM / 16);
    gemm_f16_wmma<<<dim3((tiles + 3) / 4, 1), 128, 0, stream>>>(
        ocatH, WoutT, out, L_SEQ, DN_DIM, OUT_CAT,
        OUT_CAT, OUT_CAT, DN_DIM, 0, 0, 0, 1, single, bout);
  }
}
